// MambaMixer_optimize_90958817394795
// MI455X (gfx1250) — compile-verified, matrix operand order checked by reading, not run on hardware
//
#include <hip/hip_runtime.h>
#include <hip/hip_bf16.h>
#include <math.h>

// Problem constants (B=1)
#define L_SEQ   1024
#define H_DIM   2048
#define DI_DIM  4096
#define N_STATE 16
#define R_RANK  128
#define K_CONV  4

typedef __bf16 bf16_t;
typedef __attribute__((ext_vector_type(16))) __bf16 v16bf;
typedef __attribute__((ext_vector_type(8)))  __bf16 v8bf;
typedef __attribute__((ext_vector_type(4)))  __bf16 v4bf;
typedef __attribute__((ext_vector_type(8)))  float  v8f;

// ---------------------------------------------------------------------------
// WMMA helpers (CDNA5 layouts, ISA 7.12.2):
//  A (16x32 bf16, K-major in memory): lane -> row = lane&15; K pairs:
//    VGPR0..3 hold K = kb+0..7, VGPR4..7 hold K = kb+16..23, kb = (lane<16?0:8)
//  B (32x16): symmetric with N on lanes; loaded from a K-major weight matrix
//    whose ROW index is the output column (B[k][n] = W[n][k]).
//  C/D (16x16 f32): VGPR r -> row r + 8*(lane>=16), col = lane&15
// ---------------------------------------------------------------------------
static __device__ __forceinline__ v16bf load_frag_kmaj(const bf16_t* __restrict__ base,
                                                       int ld, int row0, int k0) {
  const int lane = threadIdx.x & 31;
  const int r    = row0 + (lane & 15);
  const int kb   = k0 + ((lane >> 4) << 3);          // +8 for upper half-wave
  const bf16_t* p = base + (size_t)r * (size_t)ld + kb;
  v8bf lo = *reinterpret_cast<const v8bf*>(p);       // K = kb .. kb+7
  v8bf hi = *reinterpret_cast<const v8bf*>(p + 16);  // K = kb+16 .. kb+23
  v16bf a;
#pragma unroll
  for (int i = 0; i < 8; ++i) { a[i] = lo[i]; a[i + 8] = hi[i]; }
  return a;
}

static __device__ __forceinline__ v8f wmma_bf16(v16bf a, v16bf b, v8f c) {
  // (neg_a, A, neg_b, B, c_mod, C, reuse_a, reuse_b)
  return __builtin_amdgcn_wmma_f32_16x16x32_bf16(false, a, false, b, (short)0, c,
                                                 false, false);
}

static __device__ __forceinline__ float silu_f(float g) {
  return g / (1.0f + __expf(-g));
}

// Async global -> LDS copy, 16B per lane (GV mode: per-lane 64-bit global
// address in VGPR pair, per-lane LDS byte offset in a VGPR). Tracked by
// ASYNCcnt; completion enforced with s_wait_asynccnt.
static __device__ __forceinline__ void async_ld_b128(unsigned lds_off, const bf16_t* g) {
  asm volatile("global_load_async_to_lds_b128 %0, %1, off"
               :: "v"(lds_off), "v"(g) : "memory");
}

// ---------------------------------------------------------------------------
// fp32 -> bf16 bulk convert (4 elems/thread, all sizes are multiples of 4)
// ---------------------------------------------------------------------------
__global__ void __launch_bounds__(256) cvt_f32_bf16(const float* __restrict__ in,
                                                    bf16_t* __restrict__ out, int n4) {
  int i = blockIdx.x * blockDim.x + threadIdx.x;
  if (i >= n4) return;
  const float4 v = reinterpret_cast<const float4*>(in)[i];
  v4bf o;
  o[0] = (bf16_t)v.x; o[1] = (bf16_t)v.y; o[2] = (bf16_t)v.z; o[3] = (bf16_t)v.w;
  reinterpret_cast<v4bf*>(out)[i] = o;
}

// ---------------------------------------------------------------------------
// Generic K-major bf16 WMMA GEMM: C[m, n] = A[m,:] . B[n,:]   (ld = K both)
// Wave tile 64(M) x 32(N); block = 8 waves, 2(M) x 4(N) -> 128 x 128 tile.
// The 128x32 B tile for each K-step is staged into LDS with async
// global->LDS b128 copies, double-buffered (2 x 8 KB), so the weight stream
// is fetched once per block instead of once per wave, and the copy overlaps
// the WMMAs of the previous K-step. A fragments load direct from global.
// EPI: 0 = f32 store, 2 = bf16-only store, 3 = softplus(acc + bias[col]) f32
// ---------------------------------------------------------------------------
template <int EPI>
__global__ void __launch_bounds__(256, 1)
gemm_kmaj(const bf16_t* __restrict__ A,
          const bf16_t* __restrict__ Bw,
          float* __restrict__ Cf,
          bf16_t* __restrict__ Cbf,
          const float* __restrict__ bias,
          int N, int K) {
  __shared__ __align__(16) bf16_t sB[2][128 * 32];   // 2 stages x 8 KB
  const int wave = threadIdx.x >> 5;
  const int row0 = blockIdx.x * 128 + (wave >> 2) * 64;
  const int colL = (wave & 3) * 32;                  // col offset inside tile
  const int col0 = blockIdx.y * 128 + colL;
  const int rowB = blockIdx.y * 128;                 // first weight row of tile
  const int t    = threadIdx.x;
  const int nk   = K / 32;

  // cooperative async stage of B tile: 512 x 16B chunks, 2 per thread
  auto stage = [&](int kt, int buf) {
#pragma unroll
    for (int i = 0; i < 2; ++i) {
      const int chunk = t + i * 256;                 // 0..511
      const int r = chunk >> 2;                      // tile row 0..127
      const int c = chunk & 3;                       // 16B chunk in row
      const bf16_t* g = Bw + (size_t)(rowB + r) * (size_t)K + kt * 32 + c * 8;
      // low 32 bits of the generic shared-pointer value = LDS byte offset
      async_ld_b128((unsigned)(size_t)&sB[buf][chunk * 8], g);
    }
  };

  v8f acc[4][2] = {};
  stage(0, 0);
#pragma unroll 1
  for (int kt = 0; kt < nk; ++kt) {
    __syncthreads();                 // everyone done reading buffer (kt+1)&1
    if (kt + 1 < nk) {
      stage(kt + 1, (kt + 1) & 1);
      asm volatile("s_wait_asynccnt 0x2" ::: "memory");  // current stage landed
    } else {
      asm volatile("s_wait_asynccnt 0x0" ::: "memory");
    }
    __syncthreads();                 // stage visible to all waves

    const int k = kt * 32;
    v16bf a0 = load_frag_kmaj(A, K, row0,      k);
    v16bf a1 = load_frag_kmaj(A, K, row0 + 16, k);
    v16bf a2 = load_frag_kmaj(A, K, row0 + 32, k);
    v16bf a3 = load_frag_kmaj(A, K, row0 + 48, k);
    if (kt + 1 < nk) __builtin_prefetch(A + (size_t)row0 * K + k + 32, 0, 3);
    const bf16_t* sb = &sB[kt & 1][0];
    v16bf b0 = load_frag_kmaj(sb, 32, colL,      0);
    v16bf b1 = load_frag_kmaj(sb, 32, colL + 16, 0);

    acc[0][0] = wmma_bf16(a0, b0, acc[0][0]);
    acc[0][1] = wmma_bf16(a0, b1, acc[0][1]);
    acc[1][0] = wmma_bf16(a1, b0, acc[1][0]);
    acc[1][1] = wmma_bf16(a1, b1, acc[1][1]);
    acc[2][0] = wmma_bf16(a2, b0, acc[2][0]);
    acc[2][1] = wmma_bf16(a2, b1, acc[2][1]);
    acc[3][0] = wmma_bf16(a3, b0, acc[3][0]);
    acc[3][1] = wmma_bf16(a3, b1, acc[3][1]);
  }

  const int lane = threadIdx.x & 31;
  const int cl   = lane & 15;
  const int rb   = (lane >> 4) << 3;
#pragma unroll
  for (int i = 0; i < 4; ++i)
#pragma unroll
    for (int j = 0; j < 2; ++j) {
      const int col = col0 + j * 16 + cl;
      const int r0  = row0 + i * 16 + rb;
#pragma unroll
      for (int r = 0; r < 8; ++r) {
        float v = acc[i][j][r];
        const size_t idx = (size_t)(r0 + r) * (size_t)N + col;
        if (EPI == 3) {
          const float tt = v + bias[col];
          v = (tt > 20.0f) ? tt : __logf(1.0f + __expf(tt));  // softplus
          Cf[idx] = v;
        } else if (EPI == 0) {
          Cf[idx] = v;
        } else if (EPI == 2) {
          Cbf[idx] = (bf16_t)v;
        }
      }
    }
}

// ---------------------------------------------------------------------------
// Bm[l,n] / Cm[l,n] (N_STATE = 16 wide): one wave = 32-row tile, dual output.
// B rows (16) are tiny and L2-resident -> direct global fragment loads.
// ---------------------------------------------------------------------------
__global__ void __launch_bounds__(256, 1)
gemm_bc(const bf16_t* __restrict__ Hst,
        const bf16_t* __restrict__ Wb,
        const bf16_t* __restrict__ Wc,
        float* __restrict__ Bmat, float* __restrict__ Cmat) {
  const int wave = threadIdx.x >> 5;
  const int row0 = blockIdx.x * 256 + wave * 32;
  const int K = DI_DIM;
  v8f ab[2] = {}; v8f ac[2] = {};
#pragma unroll 1
  for (int k = 0; k < K; k += 32) {
    v16bf a0 = load_frag_kmaj(Hst, K, row0,      k);
    v16bf a1 = load_frag_kmaj(Hst, K, row0 + 16, k);
    v16bf bb = load_frag_kmaj(Wb,  K, 0,         k);
    v16bf bc = load_frag_kmaj(Wc,  K, 0,         k);
    ab[0] = wmma_bf16(a0, bb, ab[0]);
    ab[1] = wmma_bf16(a1, bb, ab[1]);
    ac[0] = wmma_bf16(a0, bc, ac[0]);
    ac[1] = wmma_bf16(a1, bc, ac[1]);
  }
  const int lane = threadIdx.x & 31;
  const int col  = lane & 15;
  const int rb   = (lane >> 4) << 3;
#pragma unroll
  for (int i = 0; i < 2; ++i) {
    const int r0 = row0 + i * 16 + rb;
#pragma unroll
    for (int r = 0; r < 8; ++r) {
      Bmat[(size_t)(r0 + r) * N_STATE + col] = ab[i][r];
      Cmat[(size_t)(r0 + r) * N_STATE + col] = ac[i][r];
    }
  }
}

// ---------------------------------------------------------------------------
// Causal depthwise conv (K=4) + bias + SiLU; outputs f32 and bf16 copies.
// ---------------------------------------------------------------------------
__global__ void __launch_bounds__(256)
conv_silu(const float* __restrict__ hs_pre,
          const float* __restrict__ conv_w,
          const float* __restrict__ conv_b,
          float* __restrict__ hst, bf16_t* __restrict__ hst_bf) {
  const int idx = blockIdx.x * blockDim.x + threadIdx.x;   // l*DI + d
  if (idx >= L_SEQ * DI_DIM) return;
  const int d = idx & (DI_DIM - 1);
  const int l = idx >> 12;                                  // DI = 4096
  float acc = conv_b[d];
#pragma unroll
  for (int j = 0; j < K_CONV; ++j) {
    const int ll = l - (K_CONV - 1) + j;
    if (ll >= 0) acc = fmaf(conv_w[d * K_CONV + j], hs_pre[(size_t)ll * DI_DIM + d], acc);
  }
  const float v = silu_f(acc);
  hst[idx]    = v;
  hst_bf[idx] = (bf16_t)v;
}

// ---------------------------------------------------------------------------
// Selective scan: one lane per (d, n) -> 65536 lanes (2048 wave32s).
// s_l = exp(A*dt)*s_{l-1} + dt*B*u ;  y = sum_n s*C  (shfl_xor over 16 lanes)
// Fused epilogue: y = (y + u*D) * silu(gate), written as bf16 for final GEMM.
// ---------------------------------------------------------------------------
__global__ void __launch_bounds__(256)
selective_scan(const float* __restrict__ dt,
               const float* __restrict__ hst,
               const float* __restrict__ gate,
               const float* __restrict__ Bmat,
               const float* __restrict__ Cmat,
               const float* __restrict__ A_log,
               const float* __restrict__ Dp,
               bf16_t* __restrict__ y_bf) {
  const int n = threadIdx.x & 15;
  const int d = blockIdx.x * 16 + (threadIdx.x >> 4);
  const float A  = -__expf(A_log[d * N_STATE + n]);
  const float Dv = Dp[d];
  float s = 0.0f;
  for (int l = 0; l < L_SEQ; ++l) {
    const float dtv = dt [(size_t)l * DI_DIM + d];   // broadcast across 16 lanes
    const float u   = hst[(size_t)l * DI_DIM + d];
    const float b   = Bmat[l * N_STATE + n];
    const float c   = Cmat[l * N_STATE + n];
    const float dA  = __expf(A * dtv);
    s = fmaf(dA, s, dtv * b * u);
    float yp = s * c;
    yp += __shfl_xor(yp, 1, 32);
    yp += __shfl_xor(yp, 2, 32);
    yp += __shfl_xor(yp, 4, 32);
    yp += __shfl_xor(yp, 8, 32);
    if (n == 0) {
      const float g  = gate[(size_t)l * DI_DIM + d];
      const float yv = (yp + u * Dv) * silu_f(g);
      y_bf[(size_t)l * DI_DIM + d] = (bf16_t)yv;
    }
  }
}

// ---------------------------------------------------------------------------
// Host-side orchestration
// ---------------------------------------------------------------------------
extern "C" void kernel_launch(void* const* d_in, const int* in_sizes, int n_in,
                              void* d_out, int out_size, void* d_ws, size_t ws_size,
                              hipStream_t stream) {
  (void)in_sizes; (void)n_in; (void)out_size; (void)ws_size;
  const float* x        = (const float*)d_in[0];
  const float* Wis      = (const float*)d_in[1];
  const float* Wig      = (const float*)d_in[2];
  const float* conv_w   = (const float*)d_in[3];
  const float* conv_b   = (const float*)d_in[4];
  const float* Wdt      = (const float*)d_in[5];
  const float* Wb       = (const float*)d_in[6];
  const float* Wc       = (const float*)d_in[7];
  const float* Wdtp     = (const float*)d_in[8];
  const float* b_dtproj = (const float*)d_in[9];
  const float* A_log    = (const float*)d_in[10];
  const float* Dp       = (const float*)d_in[11];
  const float* Wout     = (const float*)d_in[12];
  float* out = (float*)d_out;

  // workspace carve-out (256B aligned)
  char* ws = (char*)d_ws;
  size_t off = 0;
  auto alloc = [&](size_t bytes) -> void* {
    void* p = ws + off;
    off = (off + bytes + 255) & ~(size_t)255;
    return p;
  };
  bf16_t* xbf     = (bf16_t*)alloc((size_t)L_SEQ * H_DIM  * 2);
  bf16_t* wis_bf  = (bf16_t*)alloc((size_t)DI_DIM * H_DIM * 2);
  bf16_t* wig_bf  = (bf16_t*)alloc((size_t)DI_DIM * H_DIM * 2);
  bf16_t* wout_bf = (bf16_t*)alloc((size_t)H_DIM * DI_DIM * 2);
  bf16_t* wdt_bf  = (bf16_t*)alloc((size_t)R_RANK * DI_DIM * 2);
  bf16_t* wb_bf   = (bf16_t*)alloc((size_t)N_STATE * DI_DIM * 2);
  bf16_t* wc_bf   = (bf16_t*)alloc((size_t)N_STATE * DI_DIM * 2);
  bf16_t* wdtp_bf = (bf16_t*)alloc((size_t)DI_DIM * R_RANK * 2);
  float*  hs_pre  = (float*) alloc((size_t)L_SEQ * DI_DIM * 4);
  float*  gate    = (float*) alloc((size_t)L_SEQ * DI_DIM * 4);
  float*  hst     = (float*) alloc((size_t)L_SEQ * DI_DIM * 4);
  bf16_t* hst_bf  = (bf16_t*)alloc((size_t)L_SEQ * DI_DIM * 2);
  bf16_t* ts_bf   = (bf16_t*)alloc((size_t)L_SEQ * R_RANK * 2);
  float*  dt      = (float*) alloc((size_t)L_SEQ * DI_DIM * 4);
  float*  Bmat    = (float*) alloc((size_t)L_SEQ * N_STATE * 4);
  float*  Cmat    = (float*) alloc((size_t)L_SEQ * N_STATE * 4);
  bf16_t* y_bf    = (bf16_t*)alloc((size_t)L_SEQ * DI_DIM * 2);

  auto cvt = [&](const float* src, bf16_t* dst, size_t n) {
    const int n4 = (int)(n / 4);
    cvt_f32_bf16<<<(n4 + 255) / 256, 256, 0, stream>>>(src, dst, n4);
  };
  cvt(x,    xbf,     (size_t)L_SEQ * H_DIM);
  cvt(Wis,  wis_bf,  (size_t)DI_DIM * H_DIM);
  cvt(Wig,  wig_bf,  (size_t)DI_DIM * H_DIM);
  cvt(Wout, wout_bf, (size_t)H_DIM * DI_DIM);
  cvt(Wdt,  wdt_bf,  (size_t)R_RANK * DI_DIM);
  cvt(Wb,   wb_bf,   (size_t)N_STATE * DI_DIM);
  cvt(Wc,   wc_bf,   (size_t)N_STATE * DI_DIM);
  cvt(Wdtp, wdtp_bf, (size_t)DI_DIM * R_RANK);

  // 1a) state projection: [L,H] x [DI,H]^T -> hs_pre [L,DI]
  gemm_kmaj<0><<<dim3(L_SEQ / 128, DI_DIM / 128), 256, 0, stream>>>(
      xbf, wis_bf, hs_pre, (bf16_t*)nullptr, (const float*)nullptr, DI_DIM, H_DIM);
  // 1b) gate projection: [L,H] x [DI,H]^T -> gate [L,DI]
  gemm_kmaj<0><<<dim3(L_SEQ / 128, DI_DIM / 128), 256, 0, stream>>>(
      xbf, wig_bf, gate, (bf16_t*)nullptr, (const float*)nullptr, DI_DIM, H_DIM);

  // 2) causal conv + SiLU -> hst (f32 + bf16)
  conv_silu<<<(L_SEQ * DI_DIM) / 256, 256, 0, stream>>>(hs_pre, conv_w, conv_b, hst, hst_bf);

  // 3) time_step [L,R] (bf16 only, feeds dt-proj GEMM)
  gemm_kmaj<2><<<dim3(L_SEQ / 128, R_RANK / 128), 256, 0, stream>>>(
      hst_bf, wdt_bf, (float*)nullptr, ts_bf, (const float*)nullptr, R_RANK, DI_DIM);

  // 4) Bm/Cm [L,16]
  gemm_bc<<<L_SEQ / 256, 256, 0, stream>>>(hst_bf, wb_bf, wc_bf, Bmat, Cmat);

  // 5) dt [L,DI] = softplus(ts @ Wdtp^T + b)
  gemm_kmaj<3><<<dim3(L_SEQ / 128, DI_DIM / 128), 256, 0, stream>>>(
      ts_bf, wdtp_bf, dt, (bf16_t*)nullptr, b_dtproj, DI_DIM, R_RANK);

  // 6) selective scan + gating epilogue -> y_bf [L,DI]
  selective_scan<<<DI_DIM / 16, 256, 0, stream>>>(dt, hst, gate, Bmat, Cmat, A_log, Dp, y_bf);

  // 7) output projection: [L,DI] x [H,DI]^T -> out [L,H]
  gemm_kmaj<0><<<dim3(L_SEQ / 128, H_DIM / 128), 256, 0, stream>>>(
      y_bf, wout_bf, out, (bf16_t*)nullptr, (const float*)nullptr, H_DIM, DI_DIM);
}
